// GeneralisedConvolution_43104291783118
// MI455X (gfx1250) — compile-verified, matrix operand order checked by reading, not run on hardware
//
#include <hip/hip_runtime.h>
#include <hip/hip_bf16.h>

typedef __attribute__((ext_vector_type(16))) __bf16 v16bf;
typedef __attribute__((ext_vector_type(8)))  __bf16 v8bf;
typedef __attribute__((ext_vector_type(4)))  __bf16 v4bf;
typedef __attribute__((ext_vector_type(8)))  float  v8f;

#define B_ 4
#define K_ 16
#define M_ 1024
#define N_ 1024
#define P_ 1024
#define Q_ 1024
#define D_ 64

// Gather a 16-element bf16 WMMA fragment from LDS following the CDNA5
// 16-bit A/B 16x32 layout. Operands staged K-contiguous -> 2x ds_load_b128.
__device__ __forceinline__ v16bf frag_gather(const __bf16* p0, int hi) {
    v16bf f;
#pragma unroll
    for (int v = 0; v < 8; ++v) {
        int k = ((v < 4) ? (2 * v) : (16 + 2 * (v - 4))) + 8 * hi;
        f[2 * v]     = p0[k];
        f[2 * v + 1] = p0[k + 1];
    }
    return f;
}

// CDNA5 async copy: 16 bytes global -> LDS, tracked by ASYNCcnt (DMA-style
// path, no VGPR round trip).
__device__ __forceinline__ void async_copy_b128(void* lds_dst, const void* gsrc) {
    unsigned dst = (unsigned)(size_t)lds_dst;   // low 32 bits = LDS byte offset
    asm volatile("global_load_async_to_lds_b128 %0, %1, off"
                 :: "v"(dst), "v"(gsrc) : "memory");
}
__device__ __forceinline__ void async_wait0() {
    asm volatile("s_wait_asynccnt 0x0" ::: "memory");
}

// ---------------------------------------------------------------------------
// Stage 1: online softmax column stats over M for each (b,k,n).
// stats[bk][n] = {col_max, 1/col_sumexp}
// ---------------------------------------------------------------------------
__global__ void __launch_bounds__(256) k_colstats(const float* __restrict__ score,
                                                  const float* __restrict__ mask,
                                                  float* __restrict__ stats) {
    const int n  = blockIdx.x * 256 + threadIdx.x;   // gridDim.x = 4
    const int bk = blockIdx.y;                        // 64
    const float* s = score + (size_t)bk * M_ * N_;
    float mx = -INFINITY, sm = 0.0f;
    for (int m = 0; m < M_; ++m) {
        if (m + 8 < M_)
            __builtin_prefetch(s + (size_t)(m + 8) * N_ + n, 0, 1);
        float v = s[(size_t)m * N_ + n] + mask[(size_t)m * N_ + n];
        if (v > mx) { sm = sm * __expf(mx - v) + 1.0f; mx = v; }
        else        { sm += __expf(v - mx); }
    }
    stats[((size_t)bk * N_ + n) * 2 + 0] = mx;
    stats[((size_t)bk * N_ + n) * 2 + 1] = 1.0f / sm;
}

// ---------------------------------------------------------------------------
// Stage 2: weight pre-transpose to K-contiguous bf16 layouts.
//   thxT[(k*64+d)][p] = theta_x[k][p][d]   (K*D, P)
//   tht2[q][k*64+d]   = theta_y[k][q][d]   (Q, K*D)
// ---------------------------------------------------------------------------
__global__ void __launch_bounds__(256) k_prep(const float* __restrict__ thx,
                                              const float* __restrict__ ty,
                                              __bf16* __restrict__ thxT,
                                              __bf16* __restrict__ tht2) {
    int t = blockIdx.x * 256 + threadIdx.x;          // covers K_*1024*D_ = 1M
    int d = t & 63;
    int r = (t >> 6) & 1023;                         // p or q
    int k = t >> 16;
    thxT[((size_t)(k * D_ + d)) * P_ + r]    = (__bf16)thx[((size_t)k * P_ + r) * D_ + d];
    tht2[(size_t)r * (K_ * D_) + k * D_ + d] = (__bf16)ty[((size_t)k * Q_ + r) * D_ + d];
}

// ---------------------------------------------------------------------------
// Stage 3: xt2[bk][d][m] = bf16( (x[b] @ theta_x[k])[m][d] + bias_o[k][d] )
// 64 m x 64 d tile per block (4 waves), K-step 64. Double-buffered pipeline:
// theta tile via async DMA, x tile via batched float4 loads held in regs
// across the WMMA block. One barrier per iteration.
// ---------------------------------------------------------------------------
__global__ void __launch_bounds__(128) k_xt(const float* __restrict__ x,
                                            const __bf16* __restrict__ thxT,
                                            const float* __restrict__ bias_o,
                                            __bf16* __restrict__ xt2) {
    __shared__ __align__(16) __bf16 As[2][64][72];   // 64 m x 64 p
    __shared__ __align__(16) __bf16 Bt[2][64][72];   // 64 d x 64 p
    const int bk = blockIdx.y;
    const int b = bk >> 4, k = bk & 15;
    const int m0 = blockIdx.x * 64;
    const int tid = threadIdx.x;
    const int wave = tid >> 5, lane = tid & 31;
    const int col = lane & 15, hi = lane >> 4;
    v8f acc[4] = {};

    auto issue_theta = [&](int p0, int s) {
#pragma unroll
        for (int i = 0; i < 4; ++i) {               // 512 x 16B chunks
            int idx = tid + i * 128;
            int d = idx >> 3, pc = (idx & 7) * 8;
            async_copy_b128(&Bt[s][d][pc],
                            &thxT[((size_t)(k * D_ + d)) * P_ + p0 + pc]);
        }
    };
    auto load_x = [&](int p0, float4* tx) {
#pragma unroll
        for (int i = 0; i < 8; ++i) {               // 1024 float4 elems
            int idx = tid + i * 128;
            int m = idx >> 4, p4 = (idx & 15) * 4;
            tx[i] = *(const float4*)&x[((size_t)b * M_ + m0 + m) * P_ + p0 + p4];
        }
    };
    auto store_x = [&](const float4* tx, int s) {
#pragma unroll
        for (int i = 0; i < 8; ++i) {
            int idx = tid + i * 128;
            int m = idx >> 4, p4 = (idx & 15) * 4;
            v4bf w;
            w[0] = (__bf16)tx[i].x; w[1] = (__bf16)tx[i].y;
            w[2] = (__bf16)tx[i].z; w[3] = (__bf16)tx[i].w;
            *(v4bf*)&As[s][m][p4] = w;
        }
    };

    {   // prologue: stage tile 0 into buffer 0
        float4 tx[8];
        issue_theta(0, 0);
        load_x(0, tx);
        store_x(tx, 0);
    }
    int cur = 0;
    for (int t = 0; t < P_ / 64; ++t) {
        async_wait0();
        __syncthreads();                            // buf[cur] ready for all waves
        float4 tx[8];
        const bool more = (t + 1) < (P_ / 64);
        if (more) {
            issue_theta((t + 1) * 64, cur ^ 1);     // DMA flies under WMMA
            load_x((t + 1) * 64, tx);               // global loads fly under WMMA
        }
#pragma unroll
        for (int kc = 0; kc < 64; kc += 32) {
            v16bf a = frag_gather(&As[cur][wave * 16 + col][kc], hi);
#pragma unroll
            for (int dt = 0; dt < 4; ++dt) {
                v16bf bb = frag_gather(&Bt[cur][dt * 16 + col][kc], hi);
                acc[dt] = __builtin_amdgcn_wmma_f32_16x16x32_bf16(
                    false, a, false, bb, (short)0, acc[dt], false, false);
            }
        }
        if (more) store_x(tx, cur ^ 1);             // cvt+store after compute
        cur ^= 1;
    }
    const int mbase = m0 + wave * 16 + 8 * hi;      // 8 consecutive m per lane
#pragma unroll
    for (int dt = 0; dt < 4; ++dt) {
        int d = dt * 16 + col;
        float bo = bias_o[k * D_ + d];
        v8bf w;
#pragma unroll
        for (int r = 0; r < 8; ++r) w[r] = (__bf16)(acc[dt][r] + bo);
        *(v8bf*)&xt2[((size_t)(bk * D_ + d)) * M_ + mbase] = w;
    }
}

// ---------------------------------------------------------------------------
// Stage 4: zbuf[b][n][k*64+d] = bf16( sum_m attn[bk][m][n] * xt[bk][m][d] )
// Fused normalize; per-thread column stats hoisted. Double-buffered: xt tile
// via async DMA, score+mask held in regs across WMMA. 16 n x 64 d per block.
// ---------------------------------------------------------------------------
__global__ void __launch_bounds__(128) k_z(const float* __restrict__ score,
                                           const float* __restrict__ mask,
                                           const float* __restrict__ stats,
                                           const __bf16* __restrict__ xt2,
                                           __bf16* __restrict__ zbuf) {
    __shared__ __align__(16) __bf16 Ap[2][16][72];   // 16 n x 64 m : attn^T
    __shared__ __align__(16) __bf16 Bt[2][64][72];   // 64 d x 64 m : xt tile
    const int bk = blockIdx.y;
    const int b = bk >> 4, k = bk & 15;
    const int n0 = blockIdx.x * 16;
    const int tid = threadIdx.x;
    const int wave = tid >> 5, lane = tid & 31;
    const int col = lane & 15, hi = lane >> 4;
    v8f acc = {};
    const float* sc = score + (size_t)bk * M_ * N_;
    // Per-thread column (n) constant across all staging: hoist stats.
    const int nloc = tid & 15, mmb = tid >> 4;       // mm = mmb + 8*i
    const int gn = n0 + nloc;
    const float mx = stats[((size_t)bk * N_ + gn) * 2 + 0];
    const float ri = stats[((size_t)bk * N_ + gn) * 2 + 1];

    auto issue_xt = [&](int m0, int s) {
#pragma unroll
        for (int i = 0; i < 4; ++i) {
            int idx = tid + i * 128;
            int d = idx >> 3, mc = (idx & 7) * 8;
            async_copy_b128(&Bt[s][d][mc],
                            &xt2[((size_t)(bk * D_ + d)) * M_ + m0 + mc]);
        }
    };
    auto load_sm = [&](int m0, float* sv, float* mv) {
#pragma unroll
        for (int i = 0; i < 8; ++i) {
            int gm = m0 + mmb + i * 8;
            sv[i] = sc[(size_t)gm * N_ + gn];
            mv[i] = mask[(size_t)gm * N_ + gn];
        }
    };
    auto store_prob = [&](const float* sv, const float* mv, int s) {
#pragma unroll
        for (int i = 0; i < 8; ++i)
            Ap[s][nloc][mmb + i * 8] = (__bf16)(__expf(sv[i] + mv[i] - mx) * ri);
    };

    {   // prologue
        float sv[8], mv[8];
        issue_xt(0, 0);
        load_sm(0, sv, mv);
        store_prob(sv, mv, 0);
    }
    int cur = 0;
    for (int t = 0; t < M_ / 64; ++t) {
        async_wait0();
        __syncthreads();
        float sv[8], mv[8];
        const bool more = (t + 1) < (M_ / 64);
        if (more) {
            issue_xt((t + 1) * 64, cur ^ 1);
            load_sm((t + 1) * 64, sv, mv);
        }
#pragma unroll
        for (int kc = 0; kc < 64; kc += 32) {
            v16bf a  = frag_gather(&Ap[cur][col][kc], hi);
            v16bf bb = frag_gather(&Bt[cur][wave * 16 + col][kc], hi);
            acc = __builtin_amdgcn_wmma_f32_16x16x32_bf16(
                false, a, false, bb, (short)0, acc, false, false);
        }
        if (more) store_prob(sv, mv, cur ^ 1);      // exp + store after compute
        cur ^= 1;
    }
    // Epilogue: bounce accumulator tile (16n x 64d) through LDS as bf16,
    // then write coalesced 128-bit rows to zbuf (b, n, kd).
    __syncthreads();                                 // all waves done computing
    __bf16* tile = &Bt[0][0][0];                     // reuse, flat [16][64]
#pragma unroll
    for (int r = 0; r < 8; ++r)
        tile[(r + 8 * hi) * 64 + wave * 16 + col] = (__bf16)acc[r];
    __syncthreads();
    {
        int n = tid >> 3, c8 = (tid & 7) * 8;
        *(v8bf*)&zbuf[((size_t)b * N_ + n0 + n) * (K_ * D_) + k * D_ + c8] =
            *(const v8bf*)&tile[n * 64 + c8];
    }
}

// ---------------------------------------------------------------------------
// Stage 5: y[b][n][q] = zbuf[b] (N x 1024) @ tht2^T + bias_oo, fp32 out
// 64 n x 64 q per block; both tiles pure bf16 -> fully async double-buffered.
// ---------------------------------------------------------------------------
__global__ void __launch_bounds__(128) k_y(const __bf16* __restrict__ zbuf,
                                           const __bf16* __restrict__ tht2,
                                           const float* __restrict__ bias_oo,
                                           float* __restrict__ out) {
    __shared__ __align__(16) __bf16 As[2][64][72];   // 64 n x 64 kd
    __shared__ __align__(16) __bf16 Bt[2][64][72];   // 64 q x 64 kd
    const int b  = blockIdx.z;
    const int n0 = blockIdx.x * 64;
    const int q0 = blockIdx.y * 64;
    const int tid = threadIdx.x;
    const int wave = tid >> 5, lane = tid & 31;
    const int col = lane & 15, hi = lane >> 4;
    v8f acc[4] = {};

    auto issue_tiles = [&](int c0, int s) {
#pragma unroll
        for (int i = 0; i < 4; ++i) {
            int idx = tid + i * 128;
            int n = idx >> 3, cc = (idx & 7) * 8;
            async_copy_b128(&As[s][n][cc],
                            &zbuf[((size_t)b * N_ + n0 + n) * (K_ * D_) + c0 + cc]);
        }
#pragma unroll
        for (int i = 0; i < 4; ++i) {
            int idx = tid + i * 128;
            int q = idx >> 3, cc = (idx & 7) * 8;
            async_copy_b128(&Bt[s][q][cc],
                            &tht2[(size_t)(q0 + q) * (K_ * D_) + c0 + cc]);
        }
    };

    issue_tiles(0, 0);                               // prologue
    int cur = 0;
    for (int t = 0; t < (K_ * D_) / 64; ++t) {
        async_wait0();
        __syncthreads();
        if (t + 1 < (K_ * D_) / 64)
            issue_tiles((t + 1) * 64, cur ^ 1);      // DMA under WMMA
#pragma unroll
        for (int kc = 0; kc < 64; kc += 32) {
            v16bf a = frag_gather(&As[cur][wave * 16 + col][kc], hi);
#pragma unroll
            for (int qt = 0; qt < 4; ++qt) {
                v16bf bb = frag_gather(&Bt[cur][qt * 16 + col][kc], hi);
                acc[qt] = __builtin_amdgcn_wmma_f32_16x16x32_bf16(
                    false, a, false, bb, (short)0, acc[qt], false, false);
            }
        }
        cur ^= 1;
    }
#pragma unroll
    for (int qt = 0; qt < 4; ++qt)
#pragma unroll
        for (int r = 0; r < 8; ++r) {
            int n = n0 + wave * 16 + r + 8 * hi;
            int q = q0 + qt * 16 + col;
            out[((size_t)b * N_ + n) * Q_ + q] = acc[qt][r] + bias_oo[q];
        }
}

extern "C" void kernel_launch(void* const* d_in, const int* in_sizes, int n_in,
                              void* d_out, int out_size, void* d_ws, size_t ws_size,
                              hipStream_t stream) {
    (void)in_sizes; (void)n_in; (void)out_size; (void)ws_size;
    const float* score   = (const float*)d_in[0];  // (B,K,M,N)
    const float* x       = (const float*)d_in[1];  // (B,M,P)
    const float* mask    = (const float*)d_in[2];  // (M,N)
    const float* theta_x = (const float*)d_in[3];  // (K,P,D)
    const float* theta_y = (const float*)d_in[4];  // (K,Q,D)
    const float* bias_o  = (const float*)d_in[5];  // (1,K,1,D)
    const float* bias_oo = (const float*)d_in[6];  // (1,1,Q)
    float* out = (float*)d_out;                    // (B,N,Q)

    char* ws = (char*)d_ws;
    float*  stats = (float*)ws;                                           // 512 KB
    __bf16* xt2   = (__bf16*)(ws + 512u * 1024u);                         // 8 MB
    __bf16* zb    = (__bf16*)(ws + 512u * 1024u + 8u * 1024u * 1024u);    // 8 MB
    __bf16* tht2  = (__bf16*)(ws + 512u * 1024u + 16u * 1024u * 1024u);   // 2 MB
    __bf16* thxT  = (__bf16*)(ws + 512u * 1024u + 18u * 1024u * 1024u);   // 2 MB

    k_colstats<<<dim3(4, B_ * K_),       256, 0, stream>>>(score, mask, stats);
    k_prep    <<<dim3((K_*Q_*D_)/256),   256, 0, stream>>>(theta_x, theta_y, thxT, tht2);
    k_xt      <<<dim3(M_/64, B_ * K_),   128, 0, stream>>>(x, thxT, bias_o, xt2);
    k_z       <<<dim3(N_/16, B_ * K_),   128, 0, stream>>>(score, mask, stats, xt2, zb);
    k_y       <<<dim3(N_/64, Q_/64, B_), 128, 0, stream>>>(zb, tht2, bias_oo, out);
}
